// ConvexDisplacementUpdate_58583353917792
// MI455X (gfx1250) — matrix operand order repaired and checked by reading, not compiled, not run on hardware
//
#include <hip/hip_runtime.h>
#include <stdint.h>

typedef __attribute__((ext_vector_type(16))) _Float16 v16h;
typedef __attribute__((ext_vector_type(8)))  float    v8f;

#define BB 4
#define LL 4096
#define DD 256
#define INV_TEMP 10.0f
#define NQT (LL/16)     // 256 16-row tiles per batch
#define NCH (DD/32)     // 8 K-chunks of 32 for f16 WMMA
// Fixed softmax shift: logits are 10*(unit . unit) in [-10.01, 10.01], so
// exp(s - 12) in [e^-22, e^-2]: no overflow, no f32 underflow, and the
// final normalization cancels the shift exactly (softmax shift invariance).
#define FIXOFF 12.0f
#define L2E 1.442695040888963f

// workspace layout (bytes)
#define WQH_OFF 0
#define WKH_OFF ((size_t)(WQH_OFF) + (size_t)DD*DD*2)
#define QH_OFF  ((size_t)(WKH_OFF) + (size_t)DD*DD*2)
#define KH_OFF  ((size_t)(QH_OFF)  + (size_t)BB*LL*DD*2)

// ---------------------------------------------------------------------------
// Kernel 1: convert Wq/Wk (f32 row-major [e][d]) to f16
// ---------------------------------------------------------------------------
__global__ __launch_bounds__(256) void cvt_w_kernel(
    const float* __restrict__ Wq, const float* __restrict__ Wk,
    _Float16* __restrict__ wqh, _Float16* __restrict__ wkh) {
  const int i = blockIdx.x * 256 + threadIdx.x;   // 0..65535
  wqh[i] = (_Float16)Wq[i];
  wkh[i] = (_Float16)Wk[i];
}

// ---------------------------------------------------------------------------
// Kernel 2: q = l2norm(latents @ Wq^T), k = l2norm(latents @ Wk^T)
// One 16-row tile of latents per workgroup (1024 WGs, 8 waves).
// Outputs pre-swizzled: q in WMMA 16x32 f16 A-layout blobs, k in 32x16
// B-layout blobs (16 contiguous d per lane), 32 B per lane per chunk.
// ---------------------------------------------------------------------------
__global__ __launch_bounds__(256) void proj_kernel(
    const float* __restrict__ latents, const _Float16* __restrict__ wqh,
    const _Float16* __restrict__ wkh, _Float16* __restrict__ qh,
    _Float16* __restrict__ kh) {
  __shared__ alignas(32) _Float16 Ablob[NCH * 32 * 16];  // 8 KB latents tile (f16, A-layout)
  __shared__ float yq[16][DD];                           // 16 KB raw q rows (f32)
  __shared__ float yk[16][DD];                           // 16 KB raw k rows (f32)
  __shared__ float invq[16], invk[16];

  const int tile = blockIdx.x;          // 0..1023  (b = tile/256, row tile in batch = tile%256)
  const int t = threadIdx.x;
  const int w = t >> 5, l = t & 31, g = l >> 4, n16 = l & 15;

  // Stage latents tile -> f16 A-layout blobs in LDS.
  // A 16x32 f16 layout: lane holds row m=l%16; halves 0..15 = K[c*32+g*16 .. +15]
  // (contiguous, matching B).
  {
    const int c = t >> 5, lb = t & 31, gb = lb >> 4, m = lb & 15;
    const float* src = latents + ((size_t)tile * 16 + m) * DD + c * 32 + gb * 16;
    v16h a;
#pragma unroll
    for (int j = 0; j < 8; ++j) a[j] = (_Float16)src[j];
#pragma unroll
    for (int j = 0; j < 8; ++j) a[8 + j] = (_Float16)src[8 + j];
    *(v16h*)(&Ablob[(c * 32 + lb) * 16]) = a;
  }
  __syncthreads();

  // Each wave computes two 16x16 output col-tiles for q and for k.
  const v8f zero = {0.f, 0.f, 0.f, 0.f, 0.f, 0.f, 0.f, 0.f};
  v8f accq[2] = {zero, zero};
  v8f acck[2] = {zero, zero};
#pragma unroll
  for (int c = 0; c < NCH; ++c) {
    v16h a = *(const v16h*)(&Ablob[(c * 32 + l) * 16]);
#pragma unroll
    for (int ct = 0; ct < 2; ++ct) {
      const int e = (w * 2 + ct) * 16 + n16;  // output column (B-matrix N index)
      // B[K=d][N=e] = W[e][d]: 16 contiguous d per lane (K = c*32 + g*16 + j)
      v16h bq = *(const v16h*)(wqh + (size_t)e * DD + c * 32 + g * 16);
      v16h bk = *(const v16h*)(wkh + (size_t)e * DD + c * 32 + g * 16);
      accq[ct] = __builtin_amdgcn_wmma_f32_16x16x32_f16(false, a, false, bq,
                                                        (short)0, accq[ct], false, false);
      acck[ct] = __builtin_amdgcn_wmma_f32_16x16x32_f16(false, a, false, bk,
                                                        (short)0, acck[ct], false, false);
    }
  }

  // Scatter C tiles (lane holds N=e, VGPR r holds row r + 8*g) into LDS.
#pragma unroll
  for (int ct = 0; ct < 2; ++ct) {
    const int e = (w * 2 + ct) * 16 + n16;
#pragma unroll
    for (int r = 0; r < 8; ++r) {
      yq[r + 8 * g][e] = accq[ct][r];
      yk[r + 8 * g][e] = acck[ct][r];
    }
  }
  __syncthreads();

  // Row norms: 32 threads, one row each (16 q rows + 16 k rows).
  if (t < 32) {
    const int m = t & 15;
    const float* yrow = (t < 16) ? yq[m] : yk[m];
    float s = 0.f;
    for (int d = 0; d < DD; d += 4) {
      float4 v = *(const float4*)(yrow + d);
      s += v.x * v.x + v.y * v.y + v.z * v.z + v.w * v.w;
    }
    const float inv = 1.0f / fmaxf(sqrtf(s), 1e-12f);
    if (t < 16) invq[m] = inv; else invk[m] = inv;
  }
  __syncthreads();

  // Normalize + write pre-swizzled f16 blobs to global.
  {
    const int c = t >> 5, lb = t & 31, gb = lb >> 4, m = lb & 15;
    const size_t blob = ((size_t)tile * NCH + c) * 32 + lb;  // 32B per blob
    // q blob: A-layout (row = m)
    {
      const int base = c * 32 + gb * 16;
      const float s = invq[m];
      v16h a;
#pragma unroll
      for (int j = 0; j < 8; ++j) a[j] = (_Float16)(yq[m][base + j] * s);
#pragma unroll
      for (int j = 0; j < 8; ++j) a[8 + j] = (_Float16)(yq[m][base + 8 + j] * s);
      *(v16h*)(qh + blob * 16) = a;
    }
    // k blob: B-layout (key = m, K = c*32 + gb*16 + j contiguous)
    {
      const int base = c * 32 + gb * 16;
      const float s = invk[m];
      v16h bk2;
#pragma unroll
      for (int j = 0; j < 16; ++j) bk2[j] = (_Float16)(yk[m][base + j] * s);
      *(v16h*)(kh + blob * 16) = bk2;
    }
  }
}

// ---------------------------------------------------------------------------
// Kernel 3: flash attention with fixed-shift softmax.
// Grid = 128 WGs (4 batches x 32 q-blocks of 128 rows). 8 waves/WG, each
// wave owns one 16-row q tile held in 64 VGPRs. Key tiles stream through
// LDS two-at-a-time (double-buffered 16 KB pairs) via CDNA5 async
// global->LDS copies (ASYNCcnt). Two independent WMMA score chains per
// iteration so softmax VALU co-executes with XDL WMMA.
// ---------------------------------------------------------------------------
__global__ __launch_bounds__(256) void attn_kernel(
    const float* __restrict__ coords, const _Float16* __restrict__ qh,
    const _Float16* __restrict__ kh, const float* __restrict__ alpha_raw,
    float* __restrict__ out) {
  __shared__ alignas(32) _Float16 Kbuf[2][2 * NCH * 32 * 16];  // 2 x 16 KB (pair of key tiles)

  const int t = threadIdx.x;
  const int w = t >> 5;
  const int l = t & 31;
  const int g = l >> 4;
  const int n16 = l & 15;
  const int b = blockIdx.x >> 5;
  const int qblock = blockIdx.x & 31;
  const int qt = qblock * 8 + w;  // this wave's q tile (0..255)

  // Load the full q tile (16 rows x 256 d) into registers, A-layout blobs.
  v16h qreg[NCH];
  {
    const _Float16* qp = qh + (((size_t)(b * NQT + qt) * NCH) * 32 + l) * 16;
#pragma unroll
    for (int c = 0; c < NCH; ++c) qreg[c] = *(const v16h*)(qp + (size_t)c * 32 * 16);
  }

  const float araw = alpha_raw[0];
  const float alpha = 1.0f / (1.0f + __expf(-araw));
  const float2* cb = (const float2*)(coords + (size_t)b * LL * 2);

  float lrun[8], ax[8], ay[8];
#pragma unroll
  for (int r = 0; r < 8; ++r) { lrun[r] = 0.f; ax[r] = 0.f; ay[r] = 0.f; }

  // Async prefetch of a PAIR of key tiles (16 KB): thread t moves 2 blobs.
  auto prefetch_pair = [&](int p, int buf) {
    const int c = t >> 5, lb = t & 31;
#pragma unroll
    for (int u = 0; u < 2; ++u) {
      const _Float16* src =
          kh + (((size_t)(b * NQT + 2 * p + u) * NCH + c) * 32 + lb) * 16;
      unsigned lds = (unsigned)(uintptr_t)(&Kbuf[buf][((u * NCH + c) * 32 + lb) * 16]);
      asm volatile(
          "global_load_async_to_lds_b128 %0, %1, off\n\t"
          "global_load_async_to_lds_b128 %0, %1, off offset:16"
          :: "v"(lds), "v"(src) : "memory");
    }
  };

  prefetch_pair(0, 0);
  asm volatile("s_wait_asynccnt 0" ::: "memory");
  __syncthreads();

  const v8f zero = {0.f, 0.f, 0.f, 0.f, 0.f, 0.f, 0.f, 0.f};

  for (int p = 0; p < NQT / 2; ++p) {
    const int buf = p & 1;
    if (p + 1 < NQT / 2) prefetch_pair(p + 1, buf ^ 1);

    // coords of this lane's key columns (keys (2p)*16+n16 and (2p+1)*16+n16)
    const float2 kc0 = cb[p * 32 + n16];
    const float2 kc1 = cb[p * 32 + 16 + n16];

    v8f S0 = zero, S1 = zero;
#pragma unroll
    for (int c = 0; c < NCH; ++c) {
      v16h bk = *(const v16h*)(&Kbuf[buf][(c * 32 + l) * 16]);
      S0 = __builtin_amdgcn_wmma_f32_16x16x32_f16(false, qreg[c], false, bk,
                                                  (short)0, S0, false, false);
    }
#pragma unroll
    for (int c = 0; c < NCH; ++c) {
      v16h bk = *(const v16h*)(&Kbuf[buf][((NCH + c) * 32 + l) * 16]);
      S1 = __builtin_amdgcn_wmma_f32_16x16x32_f16(false, qreg[c], false, bk,
                                                  (short)0, S1, false, false);
    }

    // Fixed-shift softmax accumulation (no row max, no rescale): per row
    // (VGPR r holds row r + 8*g; the row's 16 keys live across a half-wave).
#pragma unroll
    for (int r = 0; r < 8; ++r) {
      const float s0 = fminf(fmaxf(S0[r] * INV_TEMP, -50.f), 50.f);
      const float s1 = fminf(fmaxf(S1[r] * INV_TEMP, -50.f), 50.f);
      const float p0 = __builtin_amdgcn_exp2f(fmaf(s0, L2E, -(FIXOFF * L2E)));
      const float p1 = __builtin_amdgcn_exp2f(fmaf(s1, L2E, -(FIXOFF * L2E)));
      lrun[r] += p0 + p1;
      ax[r] = fmaf(p0, kc0.x, fmaf(p1, kc1.x, ax[r]));
      ay[r] = fmaf(p0, kc0.y, fmaf(p1, kc1.y, ay[r]));
    }

    asm volatile("s_wait_asynccnt 0" ::: "memory");
    __syncthreads();
  }

  // Final 16-lane reductions + epilogue. disp = new - old.
#pragma unroll
  for (int r = 0; r < 8; ++r) {
    float ls = lrun[r], xs = ax[r], ys = ay[r];
#pragma unroll
    for (int off = 1; off < 16; off <<= 1) {
      ls += __shfl_xor(ls, off, 32);
      xs += __shfl_xor(xs, off, 32);
      ys += __shfl_xor(ys, off, 32);
    }
    if (n16 == r) {
      const int row = qt * 16 + r + 8 * g;
      const size_t idx = ((size_t)b * LL + row) * 2;
      const float qcx = coords[idx], qcy = coords[idx + 1];
      const float nx = alpha * (xs / ls) + (1.f - alpha) * qcx;
      const float ny = alpha * (ys / ls) + (1.f - alpha) * qcy;
      out[idx] = nx;
      out[idx + 1] = ny;
      out[(size_t)BB * LL * 2 + idx] = nx - qcx;
      out[(size_t)BB * LL * 2 + idx + 1] = ny - qcy;
    }
  }
}

// ---------------------------------------------------------------------------
extern "C" void kernel_launch(void* const* d_in, const int* in_sizes, int n_in,
                              void* d_out, int out_size, void* d_ws, size_t ws_size,
                              hipStream_t stream) {
  const float* latents = (const float*)d_in[0];
  const float* coords  = (const float*)d_in[1];
  const float* Wq      = (const float*)d_in[2];
  const float* Wk      = (const float*)d_in[3];
  const float* alpha_r = (const float*)d_in[4];
  float* out = (float*)d_out;

  char* ws = (char*)d_ws;
  _Float16* wqh = (_Float16*)(ws + WQH_OFF);
  _Float16* wkh = (_Float16*)(ws + WKH_OFF);
  _Float16* qh  = (_Float16*)(ws + QH_OFF);
  _Float16* kh  = (_Float16*)(ws + KH_OFF);

  cvt_w_kernel<<<(DD * DD) / 256, 256, 0, stream>>>(Wq, Wk, wqh, wkh);
  proj_kernel<<<(BB * LL) / 16, 256, 0, stream>>>(latents, wqh, wkh, qh, kh);
  attn_kernel<<<BB * (LL / 128), 256, 0, stream>>>(coords, qh, kh, alpha_r, out);
}
